// SingleHeadAttention_78434692759732
// MI455X (gfx1250) — matrix operand order repaired and checked by reading, not compiled, not run on hardware
//
#include <hip/hip_runtime.h>
#include <hip/hip_bf16.h>

#define DIMD 1024
#define SEQ  2048
#define BATCH 4
#define NTOK (BATCH * SEQ)

typedef __attribute__((ext_vector_type(16))) __bf16 v16bf;
typedef __attribute__((ext_vector_type(8)))  __bf16 v8bf;
typedef __attribute__((ext_vector_type(8)))  float  v8f;
typedef __attribute__((ext_vector_type(4)))  unsigned int v4u;
typedef __attribute__((ext_vector_type(8)))  int v8i;
typedef __attribute__((ext_vector_type(4)))  int v4i;

#define WMMA_BF16(a, b, c) \
  __builtin_amdgcn_wmma_f32_16x16x32_bf16(false, (a), false, (b), (short)0, (c), false, false)

#if defined(__has_builtin)
#if __has_builtin(__builtin_amdgcn_tensor_load_to_lds) && \
    __has_builtin(__builtin_amdgcn_s_wait_tensorcnt)
#define HAVE_TDM 1
#endif
#endif
#ifndef HAVE_TDM
#define HAVE_TDM 0
#endif

// ---------------------------------------------------------------------------
// A-fragment (16x32 bf16, row-major, leading dim ld).  Lane L -> row M=L&15,
// half h=L>>4; K = {8h..8h+7} then {16+8h..16+8h+7}: two contiguous 16B loads.
// ---------------------------------------------------------------------------
static __device__ __forceinline__ v16bf frag_a(const __bf16* base, int ld,
                                               int row0, int k0, int lane) {
  const int m = lane & 15, h = (lane >> 4) & 1;
  const __bf16* p = base + (size_t)(row0 + m) * ld + (k0 + 8 * h);
  v8bf lo = *(const v8bf*)(p);
  v8bf hi = *(const v8bf*)(p + 16);
  v16bf r;
#pragma unroll
  for (int i = 0; i < 8; ++i) { r[i] = lo[i]; r[8 + i] = hi[i]; }
  return r;
}

// ---------------------------------------------------------------------------
// B-fragment (32x16 bf16) with B[k][n] = src[n0+n][k] (row-major src, ld).
// Lane L -> col N=L&15, half h=L>>4; K = {8h..8h+7} then {16+8h..16+8h+7}.
// ---------------------------------------------------------------------------
static __device__ __forceinline__ v16bf frag_b(const __bf16* base, int ld,
                                               int n0, int k0, int lane) {
  const int n = lane & 15, h = (lane >> 4) & 1;
  const __bf16* p = base + (size_t)(n0 + n) * ld + (k0 + 8 * h);
  v8bf lo = *(const v8bf*)(p);
  v8bf hi = *(const v8bf*)(p + 16);
  v16bf r;
#pragma unroll
  for (int i = 0; i < 8; ++i) { r[i] = lo[i]; r[8 + i] = hi[i]; }
  return r;
}

// ---------------------------------------------------------------------------
// Tensor Data Mover: async 2D tile (rows x cols of bf16) global -> LDS.
// Descriptor per CDNA5 ISA ch.8; tracked with TENSORcnt (6-arg builtin).
// ---------------------------------------------------------------------------
#if HAVE_TDM
static __device__ __forceinline__ void tdm_load_2d(unsigned lds_byte_off,
                                                   const void* gptr,
                                                   unsigned cols, unsigned rows,
                                                   unsigned stride_elems) {
  unsigned long long ga = (unsigned long long)(size_t)gptr;
  v4u g0;
  g0[0] = 1u;                                   // count=1, load, user mode
  g0[1] = lds_byte_off;                         // lds_addr [63:32]
  g0[2] = (unsigned)ga;                         // global_addr lo
  g0[3] = (unsigned)((ga >> 32) & 0x01ffffffu)  // global_addr [56:32]
          | (2u << 30);                         // type = 2 ("image")
  unsigned long long q0 = 0, q1 = 0, q2 = 0, q3 = 0;
  q0 |= 1ull << 16;                                    // data_size: 2 bytes
  q0 |= ((unsigned long long)stride_elems) << 48;      // tensor_dim0 [79:48] lo
  q1 |= ((unsigned long long)stride_elems) >> 16;      //                  hi
  q1 |= (1ull << 20) << 16;                            // tensor_dim1 [111:80]
  q1 |= ((unsigned long long)(cols & 0xffffu)) << 48;  // tile_dim0 [127:112]
  q2 |= (unsigned long long)(rows & 0xffffu);          // tile_dim1 [143:128]
  q2 |= ((unsigned long long)stride_elems) << 32;      // dim0_stride [207:160]
  q3 |= ((unsigned long long)stride_elems) >> 32;      //              hi bits
  v8i g1;
  g1[0] = (int)q0; g1[1] = (int)(q0 >> 32);
  g1[2] = (int)q1; g1[3] = (int)(q1 >> 32);
  g1[4] = (int)q2; g1[5] = (int)(q2 >> 32);
  g1[6] = (int)q3; g1[7] = (int)(q3 >> 32);
  v4i z4 = {0, 0, 0, 0};
  v8i z8 = {0, 0, 0, 0, 0, 0, 0, 0};
  __builtin_amdgcn_tensor_load_to_lds(g0, g1, z4, z4, z8, 0);
}
#define TDM_WAIT() __builtin_amdgcn_s_wait_tensorcnt(0)
#else
#define TDM_WAIT()
#endif

// Fallback wave-cooperative slab copy (rows x cols bf16), 32 lanes.
static __device__ __forceinline__ void stage_slab(__bf16* dst, const __bf16* src,
                                                  int rows, int cols, int ld,
                                                  int lane) {
  const int cw = cols >> 3;
  for (int i = lane; i < rows * cw; i += 32) {
    const int r = i / cw, c = (i % cw) * 8;
    *(v8bf*)(dst + (size_t)r * cols + c) = *(const v8bf*)(src + (size_t)r * ld + c);
  }
}

// ---------------------------------------------------------------------------
// fp32 -> bf16 cast (x4 vectorized)
// ---------------------------------------------------------------------------
__global__ void cast_f32_bf16(const float* __restrict__ s, __bf16* __restrict__ d,
                              int n) {
  int i = (blockIdx.x * blockDim.x + threadIdx.x) * 4;
  if (i + 3 < n) {
#pragma unroll
    for (int j = 0; j < 4; ++j) d[i + j] = (__bf16)s[i + j];
  }
}

// ---------------------------------------------------------------------------
// QKV projection, one matrix per block (blockIdx.z: 0=Q, 1=K, 2=V-transposed).
// 128x128 block tile, TDM double-buffered 64-K slabs, 8 waves in 4(m) x 2(n),
// each wave 2x4 tiles -> acc[2][4] = 64 VGPRs (no spills).
// LDS: A 2x128x64 + W 2x128x64 = 64 KB.  grid (8, 64, 3), block 256.
// ---------------------------------------------------------------------------
__global__ __launch_bounds__(256, 1)
void qkv_proj(const __bf16* __restrict__ x,
              const __bf16* __restrict__ wq, const __bf16* __restrict__ wk,
              const __bf16* __restrict__ wv,
              const float* __restrict__ bq, const float* __restrict__ bk,
              const float* __restrict__ bv,
              __bf16* __restrict__ q, __bf16* __restrict__ k,
              __bf16* __restrict__ vt) {
  extern __shared__ char smem_raw[];
  __bf16* As = (__bf16*)smem_raw;   // [2][128][64]
  __bf16* Ws = As + 2 * 8192;       // [2][128][64]

  const int lane = threadIdx.x & 31, wave = threadIdx.x >> 5;
  const int wm = wave & 3, wn = wave >> 2;
  const int mrow0 = blockIdx.y * 128;
  const int ncol0 = blockIdx.x * 128;
  const int z = blockIdx.z;
  const __bf16* w    = (z == 0) ? wq : ((z == 1) ? wk : wv);
  const float*  bias = (z == 0) ? bq : ((z == 1) ? bk : bv);

  auto stage = [&](int ks, int buf) {
    if (wave == 0) {
      const __bf16* src = x + (size_t)mrow0 * DIMD + ks * 64;
#if HAVE_TDM
      tdm_load_2d((unsigned)(size_t)(void*)(As + buf * 8192), src, 64, 128, DIMD);
#else
      stage_slab(As + buf * 8192, src, 128, 64, DIMD, lane);
#endif
    } else if (wave == 1) {
      const __bf16* src = w + (size_t)ncol0 * DIMD + ks * 64;
#if HAVE_TDM
      tdm_load_2d((unsigned)(size_t)(void*)(Ws + buf * 8192), src, 64, 128, DIMD);
#else
      stage_slab(Ws + buf * 8192, src, 128, 64, DIMD, lane);
#endif
    }
  };

  v8f acc[2][4] = {};
  stage(0, 0);
  for (int ks = 0; ks < DIMD / 64; ++ks) {
    const int buf = ks & 1;
    if (wave < 2) TDM_WAIT();
    __syncthreads();                 // slab ks ready; prev buffer free
    if (ks + 1 < DIMD / 64) stage(ks + 1, buf ^ 1);

    const __bf16* Ab = As + buf * 8192;
    const __bf16* Wb = Ws + buf * 8192;
#pragma unroll
    for (int kk = 0; kk < 64; kk += 32) {
      v16bf af[2];
#pragma unroll
      for (int mi = 0; mi < 2; ++mi)
        af[mi] = frag_a(Ab, 64, wm * 32 + mi * 16, kk, lane);
#pragma unroll
      for (int nt = 0; nt < 4; ++nt) {
        v16bf bf = frag_b(Wb, 64, wn * 64 + nt * 16, kk, lane);
#pragma unroll
        for (int mi = 0; mi < 2; ++mi)
          acc[mi][nt] = WMMA_BF16(af[mi], bf, acc[mi][nt]);
      }
    }
    __syncthreads();                 // compute done before next overwrite
  }

  const int h = (lane >> 4) & 1, nc = lane & 15;
#pragma unroll
  for (int nt = 0; nt < 4; ++nt) {
    const int col = ncol0 + wn * 64 + nt * 16 + nc;
    const float fb = bias[col];
#pragma unroll
    for (int mi = 0; mi < 2; ++mi) {
#pragma unroll
      for (int r = 0; r < 8; ++r) {
        const int m = mrow0 + wm * 32 + mi * 16 + r + 8 * h;  // token index
        const __bf16 val = (__bf16)(acc[mi][nt][r] + fb);
        if (z == 0) {
          q[(size_t)m * DIMD + col] = val;
        } else if (z == 1) {
          k[(size_t)m * DIMD + col] = val;
        } else {
          const int b = m >> 11, s = m & (SEQ - 1);
          vt[((size_t)b * DIMD + col) * SEQ + s] = val;   // V stored [B][D][S]
        }
      }
    }
  }
}

// ---------------------------------------------------------------------------
// Fused attention strip, one workgroup (8 waves) per (batch, 16 query rows).
// Q strip TDM-staged to LDS (32 KB, reused by every j-tile); scores strip in
// LDS (128 KB); softmax scratch overlays the Q strip (dead after phase A).
// Total dynamic LDS = 160 KB.
// ---------------------------------------------------------------------------
__global__ __launch_bounds__(256, 1)
void attn_strip(const __bf16* __restrict__ q, const __bf16* __restrict__ k,
                const __bf16* __restrict__ vt, __bf16* __restrict__ o) {
  extern __shared__ char smem_raw[];
  float*  sc  = (float*)smem_raw;                  // [16][SEQ]
  __bf16* qs  = (__bf16*)(smem_raw + 16 * SEQ * 4);// [16][DIMD] (phase A only)
  float*  red = (float*)qs;                        // [16][16]   (phase B only)

  const int lane = threadIdx.x & 31;
  const int wave = threadIdx.x >> 5;
  const int qt = blockIdx.x;                 // 0..127
  const int b  = blockIdx.y;                 // 0..3
  const int qrow0 = b * SEQ + qt * 16;
  const float scale = 0.03125f;              // 1/sqrt(1024)

  // ---- stage Q strip (16 x 1024 bf16) into LDS ----
  {
    const __bf16* qsrc = q + (size_t)qrow0 * DIMD;
#if HAVE_TDM
    if (wave == 0) {
      tdm_load_2d((unsigned)(size_t)(void*)qs, qsrc, DIMD, 16, DIMD);
      TDM_WAIT();
    }
#else
    const int cw = DIMD >> 3;
    for (int i = threadIdx.x; i < 16 * cw; i += 256) {
      const int r = i / cw, c = (i % cw) * 8;
      *(v8bf*)(qs + r * DIMD + c) = *(const v8bf*)(qsrc + (size_t)r * DIMD + c);
    }
#endif
  }
  __syncthreads();

  // ---- Phase A: block-causal scores into LDS ----
  for (int j = wave; j <= qt; j += 8) {
    v8f acc = {};
    const int krow0 = b * SEQ + j * 16;
    for (int kk = 0; kk < DIMD; kk += 32) {
      v16bf a  = frag_a(qs, DIMD, 0, kk, lane);       // from LDS
      v16bf bb = frag_b(k, DIMD, krow0, kk, lane);    // streamed from global
      acc = WMMA_BF16(a, bb, acc);
    }
    const int h = (lane >> 4) & 1, nc = lane & 15;
#pragma unroll
    for (int r = 0; r < 8; ++r) {
      const int m = r + 8 * h;
      const int gq = qt * 16 + m, gk = j * 16 + nc;
      float v = acc[r] * scale;
      if (gk > gq) v = -__builtin_inff();             // causal mask
      sc[m * SEQ + j * 16 + nc] = v;
    }
  }
  __syncthreads();

  // ---- Phase B: softmax (exact, two-pass in LDS) ----
  const int ncols = (qt + 1) * 16;
  const int ncap  = (ncols + 31) & ~31;
  {
    const int r = threadIdx.x >> 4, seg = threadIdx.x & 15;
    float pmax = -__builtin_inff();
    for (int c = seg; c < ncols; c += 16) pmax = fmaxf(pmax, sc[r * SEQ + c]);
    red[r * 16 + seg] = pmax;
    __syncthreads();
    float rmax = -__builtin_inff();
#pragma unroll
    for (int i = 0; i < 16; ++i) rmax = fmaxf(rmax, red[r * 16 + i]);
    __syncthreads();
    float psum = 0.f;
    for (int c = seg; c < ncols; c += 16) {
      float e = __expf(sc[r * SEQ + c] - rmax);
      sc[r * SEQ + c] = e;
      psum += e;
    }
    red[r * 16 + seg] = psum;
    __syncthreads();
    float rsum = 0.f;
#pragma unroll
    for (int i = 0; i < 16; ++i) rsum += red[r * 16 + i];
    const float inv = 1.f / rsum;
    for (int c = seg; c < ncols; c += 16) sc[r * SEQ + c] *= inv;
    if (ncols + seg < ncap) sc[r * SEQ + ncols + seg] = 0.f;  // pad to K-step
  }
  __syncthreads();

  // ---- Phase C: O = P @ V (A from LDS, B from Vt, 8 d-tiles per wave) ----
  const __bf16* vtb = vt + (size_t)b * DIMD * SEQ;
  v8f acc[8] = {};
  const int m = lane & 15, h = (lane >> 4) & 1;
  for (int kk = 0; kk < ncap; kk += 32) {
    v16bf a;
#pragma unroll
    for (int i = 0; i < 8; ++i) {
      a[i]     = (__bf16)sc[m * SEQ + kk + 8 * h + i];
      a[8 + i] = (__bf16)sc[m * SEQ + kk + 16 + 8 * h + i];
    }
#pragma unroll
    for (int t = 0; t < 8; ++t) {
      const int n0 = (wave + t * 8) * 16;
      v16bf bb = frag_b(vtb, SEQ, n0, kk, lane);
      acc[t] = WMMA_BF16(a, bb, acc[t]);
    }
  }
#pragma unroll
  for (int t = 0; t < 8; ++t) {
    const int n0 = (wave + t * 8) * 16;
#pragma unroll
    for (int r = 0; r < 8; ++r) {
      const int mm = qrow0 + r + 8 * h;
      o[(size_t)mm * DIMD + n0 + (lane & 15)] = (__bf16)acc[t][r];
    }
  }
}

// ---------------------------------------------------------------------------
// Output projection: out = attn @ Wo.T + bo (fp32), 128x128 block tile,
// TDM double-buffered slabs (64 KB LDS).
// ---------------------------------------------------------------------------
__global__ __launch_bounds__(256, 1)
void out_proj(const __bf16* __restrict__ a, const __bf16* __restrict__ wo,
              const float* __restrict__ bo, float* __restrict__ out) {
  extern __shared__ char smem_raw[];
  __bf16* As = (__bf16*)smem_raw;   // [2][128][64]
  __bf16* Ws = As + 2 * 8192;       // [2][128][64]

  const int lane = threadIdx.x & 31, wave = threadIdx.x >> 5;
  const int wm = wave & 3, wn = wave >> 2;
  const int mrow0 = blockIdx.y * 128;
  const int ncol0 = blockIdx.x * 128;

  auto stage = [&](int ks, int buf) {
    if (wave == 0) {
      const __bf16* src = a + (size_t)mrow0 * DIMD + ks * 64;
#if HAVE_TDM
      tdm_load_2d((unsigned)(size_t)(void*)(As + buf * 8192), src, 64, 128, DIMD);
#else
      stage_slab(As + buf * 8192, src, 128, 64, DIMD, lane);
#endif
    } else if (wave == 1) {
      const __bf16* src = wo + (size_t)ncol0 * DIMD + ks * 64;
#if HAVE_TDM
      tdm_load_2d((unsigned)(size_t)(void*)(Ws + buf * 8192), src, 64, 128, DIMD);
#else
      stage_slab(Ws + buf * 8192, src, 128, 64, DIMD, lane);
#endif
    }
  };

  v8f acc[2][4] = {};
  stage(0, 0);
  for (int ks = 0; ks < DIMD / 64; ++ks) {
    const int buf = ks & 1;
    if (wave < 2) TDM_WAIT();
    __syncthreads();
    if (ks + 1 < DIMD / 64) stage(ks + 1, buf ^ 1);

    const __bf16* Ab = As + buf * 8192;
    const __bf16* Wb = Ws + buf * 8192;
#pragma unroll
    for (int kk = 0; kk < 64; kk += 32) {
      v16bf af[2];
#pragma unroll
      for (int mi = 0; mi < 2; ++mi)
        af[mi] = frag_a(Ab, 64, wm * 32 + mi * 16, kk, lane);
#pragma unroll
      for (int nt = 0; nt < 4; ++nt) {
        v16bf bf = frag_b(Wb, 64, wn * 64 + nt * 16, kk, lane);
#pragma unroll
        for (int mi = 0; mi < 2; ++mi)
          acc[mi][nt] = WMMA_BF16(af[mi], bf, acc[mi][nt]);
      }
    }
    __syncthreads();
  }

  const int h = (lane >> 4) & 1, nc = lane & 15;
#pragma unroll
  for (int nt = 0; nt < 4; ++nt) {
    const int col = ncol0 + wn * 64 + nt * 16 + nc;
    const float fb = bo[col];
#pragma unroll
    for (int mi = 0; mi < 2; ++mi) {
#pragma unroll
      for (int r = 0; r < 8; ++r) {
        const int m = mrow0 + wm * 32 + mi * 16 + r + 8 * h;
        out[(size_t)m * DIMD + col] = acc[mi][nt][r] + fb;
      }
    }
  }
}

// ---------------------------------------------------------------------------
extern "C" void kernel_launch(void* const* d_in, const int* in_sizes, int n_in,
                              void* d_out, int out_size, void* d_ws, size_t ws_size,
                              hipStream_t stream) {
  (void)in_sizes; (void)n_in; (void)out_size; (void)ws_size;
  const float* x  = (const float*)d_in[0];
  const float* Wq = (const float*)d_in[1];
  const float* bq = (const float*)d_in[2];
  const float* Wk = (const float*)d_in[3];
  const float* bk = (const float*)d_in[4];
  const float* Wv = (const float*)d_in[5];
  const float* bv = (const float*)d_in[6];
  const float* Wo = (const float*)d_in[7];
  const float* bo = (const float*)d_in[8];
  float* out = (float*)d_out;

  // Workspace carve-up (88 MB total)
  char* ws = (char*)d_ws;
  __bf16* xbf  = (__bf16*)ws; ws += (size_t)NTOK * DIMD * 2;  // 16 MB
  __bf16* wqbf = (__bf16*)ws; ws += (size_t)DIMD * DIMD * 2;  //  2 MB
  __bf16* wkbf = (__bf16*)ws; ws += (size_t)DIMD * DIMD * 2;
  __bf16* wvbf = (__bf16*)ws; ws += (size_t)DIMD * DIMD * 2;
  __bf16* wobf = (__bf16*)ws; ws += (size_t)DIMD * DIMD * 2;
  __bf16* qbf  = (__bf16*)ws; ws += (size_t)NTOK * DIMD * 2;  // 16 MB
  __bf16* kbf  = (__bf16*)ws; ws += (size_t)NTOK * DIMD * 2;
  __bf16* vtbf = (__bf16*)ws; ws += (size_t)NTOK * DIMD * 2;  // [B][D][S]
  __bf16* abf  = (__bf16*)ws; ws += (size_t)NTOK * DIMD * 2;  // attn out

  const int nx = NTOK * DIMD;   // 8,388,608
  const int nw = DIMD * DIMD;   // 1,048,576
  cast_f32_bf16<<<nx / 4 / 256, 256, 0, stream>>>(x,  xbf,  nx);
  cast_f32_bf16<<<nw / 4 / 256, 256, 0, stream>>>(Wq, wqbf, nw);
  cast_f32_bf16<<<nw / 4 / 256, 256, 0, stream>>>(Wk, wkbf, nw);
  cast_f32_bf16<<<nw / 4 / 256, 256, 0, stream>>>(Wv, wvbf, nw);
  cast_f32_bf16<<<nw / 4 / 256, 256, 0, stream>>>(Wo, wobf, nw);

  qkv_proj<<<dim3(DIMD / 128, NTOK / 128, 3), 256, 65536, stream>>>(
      xbf, wqbf, wkbf, wvbf, bq, bk, bv, qbf, kbf, vtbf);

  attn_strip<<<dim3(SEQ / 16, BATCH), 256, 163840, stream>>>(qbf, kbf, vtbf, abf);

  out_proj<<<dim3(DIMD / 128, NTOK / 128), 256, 65536, stream>>>(abf, wobf, bo, out);
}
